// TransformerLayer_14980845928553
// MI455X (gfx1250) — matrix operand order repaired and checked by reading, not compiled, not run on hardware
//
#include <hip/hip_runtime.h>
#include <math.h>
#include <stdint.h>

// ---------------------------------------------------------------------------
// Types for CDNA5 WMMA (wave32, v_wmma_f32_16x16x32_bf16)
// ---------------------------------------------------------------------------
typedef __attribute__((ext_vector_type(16))) __bf16 bf16x16;
typedef __attribute__((ext_vector_type(8)))  float  v8f;

#define DEV static __device__ __forceinline__

constexpr int cB = 4, cS = 2048, cD = 1024, cH = 16, cFF = 4096, cHD = 64;
constexpr size_t cM = (size_t)cB * cS;  // 8192 token rows

// ---------------------------------------------------------------------------
// Async global->LDS staging (gfx1250): GLOBAL_LOAD_ASYNC_TO_LDS_B128,
// tracked with ASYNCcnt. Generic LDS pointers keep the LDS byte offset in the
// low 32 bits (ISA 10.2), which is what VDST wants.
// ---------------------------------------------------------------------------
DEV void async_copy_b128(const void* gsrc, void* lds_dst) {
  uint32_t lds = (uint32_t)(uintptr_t)lds_dst;
  asm volatile("global_load_async_to_lds_b128 %0, %1, off"
               :
               : "v"(lds), "v"(gsrc)
               : "memory");
}
DEV void wait_async() { asm volatile("s_wait_asynccnt 0x0" ::: "memory"); }

// ---------------------------------------------------------------------------
// WMMA fragment gathers from LDS, per ISA 7.12.2 layouts (wave32).
// A (16x32 bf16, MxK): half=lane>>4, m=lane&15, k(i) = (i&7)+half*8+(i>>3)*16
// B (32x16 bf16, KxN): n=lane&15, k(i) = (i&7)+(lane>>4)*8+(i>>3)*16  (LDS holds [n][k])
// C (16x16 f32):       n=lane&15, m(r) = r + (lane>>4)*8
// ---------------------------------------------------------------------------
DEV bf16x16 frag_a(const __bf16* lds, int row0, int ld, int koff) {
  int lane = threadIdx.x & 31;
  int half = lane >> 4, m = lane & 15;
  bf16x16 f;
#pragma unroll
  for (int i = 0; i < 16; ++i) {
    int k = (i & 7) + half * 8 + ((i >> 3) << 4);
    f[i] = lds[(row0 + m) * ld + koff + k];
  }
  return f;
}

DEV bf16x16 frag_b(const __bf16* lds_t, int col0, int ld, int koff) {
  int lane = threadIdx.x & 31;
  int kh = (lane >> 4) << 3, n = lane & 15;
  bf16x16 f;
#pragma unroll
  for (int i = 0; i < 16; ++i) f[i] = lds_t[(col0 + n) * ld + koff + kh + (i & 7) + ((i >> 3) << 4)];
  return f;
}

DEV v8f wmma_bf16(bf16x16 a, bf16x16 b, v8f c) {
  return __builtin_amdgcn_wmma_f32_16x16x32_bf16(false, a, false, b, (short)0, c,
                                                 false, false);
}

DEV float gelu_tanh(float z) {
  float u = 0.7978845608028654f * (z + 0.044715f * z * z * z);
  return 0.5f * z * (1.0f + tanhf(u));
}

// ---------------------------------------------------------------------------
// Transpose-cast: src[K][N] f32  ->  dst[N][K] bf16 (32x32 LDS tiles).
// Pre-transposed weights make GEMM B-tiles K-contiguous => async-copyable.
// ---------------------------------------------------------------------------
__global__ __launch_bounds__(256) void cast_transpose_kernel(
    const float* __restrict__ src, __bf16* __restrict__ dst, int K, int N) {
  __shared__ __bf16 t[32][33];
  int k0 = blockIdx.y * 32, n0 = blockIdx.x * 32;
  int tx = threadIdx.x & 31, ty = threadIdx.x >> 5;  // 8 row-groups
  for (int r = ty; r < 32; r += 8)
    t[r][tx] = (__bf16)src[(size_t)(k0 + r) * N + n0 + tx];
  __syncthreads();
  for (int r = ty; r < 32; r += 8)
    dst[(size_t)(n0 + r) * K + k0 + tx] = t[tx][r];
}

// ---------------------------------------------------------------------------
// LayerNorm over D=1024, one 256-thread block per row; bf16 output
// ---------------------------------------------------------------------------
__global__ __launch_bounds__(256) void layernorm_kernel(
    const float* __restrict__ x, const float* __restrict__ g,
    const float* __restrict__ b, __bf16* __restrict__ out) {
  int row = blockIdx.x;
  const float* xr = x + (size_t)row * cD;
  float s = 0.f, ss = 0.f;
  for (int i = threadIdx.x; i < cD; i += 256) {
    float v = xr[i];
    s += v;
    ss += v * v;
  }
  __shared__ float rs[256], rq[256];
  rs[threadIdx.x] = s;
  rq[threadIdx.x] = ss;
  __syncthreads();
  for (int off = 128; off > 0; off >>= 1) {
    if (threadIdx.x < off) {
      rs[threadIdx.x] += rs[threadIdx.x + off];
      rq[threadIdx.x] += rq[threadIdx.x + off];
    }
    __syncthreads();
  }
  float mean = rs[0] * (1.0f / cD);
  float var = rq[0] * (1.0f / cD) - mean * mean;
  float inv = rsqrtf(var + 1e-5f);
  for (int i = threadIdx.x; i < cD; i += 256) {
    float v = (xr[i] - mean) * inv * g[i] + b[i];
    out[(size_t)row * cD + i] = (__bf16)v;
  }
}

// ---------------------------------------------------------------------------
// BF16 GEMM: C[M,N] = A[M,K] @ Bt[N,K]^T (+bias)(+GELU)(+residual).
// Block tile 128x128, BK=32, 8 waves in a 4(row)x2(col) grid:
// each wave = 2 A-frags x 4 B-frags = 8 WMMA per K-step.
// DOUBLE-BUFFERED async pipeline: tile kt+1 is issued (async, no VGPR data
// path) right after the barrier and runs concurrently with tile kt's WMMAs;
// s_wait_asynccnt 0 + barrier at loop top publish it to all waves.
// PERM: 0 = row-major [M,N]; 1 = QKV [B,H,S,HD]; 2 = V^T [B,H,HD,S]
// ---------------------------------------------------------------------------
template <bool GELU, int PERM>
__global__ __launch_bounds__(256) void gemm_bf16_kernel(
    const __bf16* __restrict__ A, const __bf16* __restrict__ Bt,
    const float* __restrict__ bias, const float* __restrict__ residual,
    __bf16* __restrict__ out_bf, float* __restrict__ out_f, int M, int N,
    int K) {
  constexpr int BM = 128, BN = 128, BK = 32;
  __shared__ __align__(16) __bf16 As[2][BM * BK];  // [m][k]
  __shared__ __align__(16) __bf16 Bs[2][BN * BK];  // [n][k]
  int tid = threadIdx.x, wave = tid >> 5, lane = tid & 31;
  int wr = wave & 3, wc = wave >> 2;
  int m0 = blockIdx.y * BM, n0 = blockIdx.x * BN;

  // stage one 128x32 tile pair = 512 x 16B chunks each; 2 chunks/thread each
  auto stage = [&](int kt, int pb) {
    int k0 = kt * BK;
#pragma unroll
    for (int it = 0; it < 2; ++it) {
      int c = tid + it * 256;
      int r = c >> 2, col = (c & 3) << 3;
      async_copy_b128(&A[(size_t)(m0 + r) * K + k0 + col],
                      &As[pb][r * BK + col]);
      async_copy_b128(&Bt[(size_t)(n0 + r) * K + k0 + col],
                      &Bs[pb][r * BK + col]);
    }
  };

  v8f acc[2][4];
  v8f z = {0.f, 0.f, 0.f, 0.f, 0.f, 0.f, 0.f, 0.f};
#pragma unroll
  for (int j = 0; j < 2; ++j)
#pragma unroll
    for (int t = 0; t < 4; ++t) acc[j][t] = z;

  int nk = K / BK;
  stage(0, 0);
  for (int kt = 0; kt < nk; ++kt) {
    int pb = kt & 1;
    wait_async();     // tile kt is the only outstanding async batch
    __syncthreads();  // publish tile kt; also fences reads of buffer pb^1
    if (kt + 1 < nk) stage(kt + 1, pb ^ 1);  // overlaps with compute below
    // preload all fragments, then back-to-back WMMAs
    bf16x16 af0 = frag_a(As[pb], wr * 32, BK, 0);
    bf16x16 af1 = frag_a(As[pb], wr * 32 + 16, BK, 0);
    bf16x16 bfr[4];
#pragma unroll
    for (int t = 0; t < 4; ++t) bfr[t] = frag_b(Bs[pb], wc * 64 + t * 16, BK, 0);
#pragma unroll
    for (int t = 0; t < 4; ++t) {
      acc[0][t] = wmma_bf16(af0, bfr[t], acc[0][t]);
      acc[1][t] = wmma_bf16(af1, bfr[t], acc[1][t]);
    }
  }

  int half = lane >> 4, nn = lane & 15;
#pragma unroll
  for (int j = 0; j < 2; ++j)
#pragma unroll
    for (int t = 0; t < 4; ++t)
#pragma unroll
      for (int r = 0; r < 8; ++r) {
        int m = m0 + wr * 32 + j * 16 + half * 8 + r;
        int n = n0 + wc * 64 + t * 16 + nn;
        float v = acc[j][t][r];
        if (bias) v += bias[n];
        if (GELU) v = gelu_tanh(v);
        if (residual) v += residual[(size_t)m * N + n];
        size_t oidx;
        if (PERM == 1) {
          int b_ = m >> 11, s_ = m & 2047;  // m = b*S+s
          int h_ = n >> 6, hd = n & 63;     // n = h*HD+hd
          oidx = (((size_t)b_ * cH + h_) * cS + s_) * cHD + hd;
        } else if (PERM == 2) {
          int b_ = m >> 11, s_ = m & 2047;
          int h_ = n >> 6, hd = n & 63;
          oidx = (((size_t)b_ * cH + h_) * cHD + hd) * cS + s_;
        } else {
          oidx = (size_t)m * N + n;
        }
        if (out_f) out_f[oidx] = v;
        if (out_bf) out_bf[oidx] = (__bf16)v;
      }
}

// ---------------------------------------------------------------------------
// scores[b,h] = (Q[b,h] @ K[b,h]^T) / 8, causal-masked, f32 out.
// Q,K in [B,H,S,HD] bf16 (HD innermost => both tiles async-copyable).
// ---------------------------------------------------------------------------
__global__ __launch_bounds__(256) void attn_scores_kernel(
    const __bf16* __restrict__ q, const __bf16* __restrict__ kmat,
    float* __restrict__ attn) {
  constexpr int BM = 128, BN = 64;
  __shared__ __align__(16) __bf16 Qs[BM * 64];  // [m][hd]
  __shared__ __align__(16) __bf16 Ks[BN * 64];  // [n=s_k][hd]
  int bh = blockIdx.z;
  int tid = threadIdx.x, wave = tid >> 5, lane = tid & 31;
  int m0 = blockIdx.y * BM, n0 = blockIdx.x * BN;
  float* ab = attn + (size_t)bh * cS * cS;
  int half = lane >> 4, nn = lane & 15;

  if (n0 > m0 + BM - 1) {  // tile entirely above diagonal: pure -inf
#pragma unroll
    for (int t = 0; t < 4; ++t)
#pragma unroll
      for (int r = 0; r < 8; ++r) {
        int sq = m0 + wave * 16 + half * 8 + r;
        int sk = n0 + t * 16 + nn;
        ab[(size_t)sq * cS + sk] = -__builtin_inff();
      }
    return;
  }

  const __bf16* qb = q + (size_t)bh * cS * cHD;
  const __bf16* kb = kmat + (size_t)bh * cS * cHD;
  // Qs: 1024 x 16B chunks (4/thread); Ks: 512 chunks (2/thread)
#pragma unroll
  for (int it = 0; it < 4; ++it) {
    int c = tid + it * 256;
    int r = c >> 3, col = (c & 7) << 3;
    async_copy_b128(&qb[(size_t)(m0 + r) * 64 + col], &Qs[r * 64 + col]);
  }
#pragma unroll
  for (int it = 0; it < 2; ++it) {
    int c = tid + it * 256;
    int r = c >> 3, col = (c & 7) << 3;
    async_copy_b128(&kb[(size_t)(n0 + r) * 64 + col], &Ks[r * 64 + col]);
  }
  wait_async();
  __syncthreads();

  v8f acc[4];
  v8f z = {0.f, 0.f, 0.f, 0.f, 0.f, 0.f, 0.f, 0.f};
#pragma unroll
  for (int t = 0; t < 4; ++t) acc[t] = z;
#pragma unroll
  for (int kk = 0; kk < 64; kk += 32) {
    bf16x16 af = frag_a(Qs, wave * 16, 64, kk);
    bf16x16 bfr[4];
#pragma unroll
    for (int t = 0; t < 4; ++t) bfr[t] = frag_b(Ks, t * 16, 64, kk);
#pragma unroll
    for (int t = 0; t < 4; ++t) acc[t] = wmma_bf16(af, bfr[t], acc[t]);
  }
#pragma unroll
  for (int t = 0; t < 4; ++t)
#pragma unroll
    for (int r = 0; r < 8; ++r) {
      int sq = m0 + wave * 16 + half * 8 + r;
      int sk = n0 + t * 16 + nn;
      float v = (sk > sq) ? -__builtin_inff() : acc[t][r] * 0.125f;
      ab[(size_t)sq * cS + sk] = v;
    }
}

// ---------------------------------------------------------------------------
// Row softmax over S=2048; one block per (b,h,s_q) row; values in registers.
// ---------------------------------------------------------------------------
__global__ __launch_bounds__(256) void softmax_kernel(float* __restrict__ attn) {
  float* p = attn + (size_t)blockIdx.x * cS;
  int tid = threadIdx.x;
  float vals[8];
  float mx = -__builtin_inff();
#pragma unroll
  for (int i = 0; i < 8; ++i) {
    vals[i] = p[i * 256 + tid];
    mx = fmaxf(mx, vals[i]);
  }
  __shared__ float red[256];
  red[tid] = mx;
  __syncthreads();
  for (int off = 128; off > 0; off >>= 1) {
    if (tid < off) red[tid] = fmaxf(red[tid], red[tid + off]);
    __syncthreads();
  }
  float m = red[0];
  __syncthreads();
  float s = 0.f;
#pragma unroll
  for (int i = 0; i < 8; ++i) {
    vals[i] = expf(vals[i] - m);  // exp(-inf - m) = 0 for masked entries
    s += vals[i];
  }
  red[tid] = s;
  __syncthreads();
  for (int off = 128; off > 0; off >>= 1) {
    if (tid < off) red[tid] += red[tid + off];
    __syncthreads();
  }
  float inv = 1.0f / red[0];
#pragma unroll
  for (int i = 0; i < 8; ++i) p[i * 256 + tid] = vals[i] * inv;
}

// ---------------------------------------------------------------------------
// O[b,h] = attn[b,h] @ V[b,h].  attn f32 -> bf16 on LDS stage (VALU path);
// V^T stored [B,H,HD,S] so the V tile is K-contiguous => async staged.
// K-blocks above the diagonal are exactly zero after softmax: skipped.
// Output stored directly in [B,S,D] layout.
// ---------------------------------------------------------------------------
__global__ __launch_bounds__(256) void attn_v_kernel(
    const float* __restrict__ attn, const __bf16* __restrict__ vt,
    __bf16* __restrict__ o) {
  constexpr int BM = 128, BK = 32;
  __shared__ __align__(16) __bf16 Ps[BM * BK];  // [m][k]
  __shared__ __align__(16) __bf16 Vs[64 * BK];  // [n=hd][k]
  int bh = blockIdx.z;
  int b_ = bh >> 4, h_ = bh & 15;
  const float* ab = attn + (size_t)bh * cS * cS;
  const __bf16* vb = vt + (size_t)bh * cHD * cS;  // [hd][s]
  int tid = threadIdx.x, wave = tid >> 5, lane = tid & 31;
  int m0 = blockIdx.y * BM;

  v8f acc[4];
  v8f z = {0.f, 0.f, 0.f, 0.f, 0.f, 0.f, 0.f, 0.f};
#pragma unroll
  for (int t = 0; t < 4; ++t) acc[t] = z;

  int kmax = m0 + BM;  // causal: attn is zero for s_k > s_q
  for (int k0 = 0; k0 < kmax; k0 += BK) {
    __syncthreads();
    {  // Vs: 64x32 bf16 = 256 x 16B chunks, 1/thread, async
      int r = tid >> 2, col = (tid & 3) << 3;
      async_copy_b128(&vb[(size_t)r * cS + k0 + col], &Vs[r * BK + col]);
    }
    // Ps: 128x32, f32 -> bf16, packed dword stores
    for (int idx = tid; idx < BM * BK / 2; idx += 256) {
      int r = idx >> 4, c = (idx & 15) << 1;
      const float* sp = &ab[(size_t)(m0 + r) * cS + k0 + c];
      union { uint32_t u; __bf16 h[2]; } cv;
      cv.h[0] = (__bf16)sp[0];
      cv.h[1] = (__bf16)sp[1];
      *(uint32_t*)&Ps[r * BK + c] = cv.u;
    }
    wait_async();
    __syncthreads();
    bf16x16 af = frag_a(Ps, wave * 16, BK, 0);
    bf16x16 bfr[4];
#pragma unroll
    for (int t = 0; t < 4; ++t) bfr[t] = frag_b(Vs, t * 16, BK, 0);
#pragma unroll
    for (int t = 0; t < 4; ++t) acc[t] = wmma_bf16(af, bfr[t], acc[t]);
  }

  int half = lane >> 4, nn = lane & 15;
#pragma unroll
  for (int t = 0; t < 4; ++t)
#pragma unroll
    for (int r = 0; r < 8; ++r) {
      int s_ = m0 + wave * 16 + half * 8 + r;
      int n = t * 16 + nn;  // hd
      o[((size_t)b_ * cS + s_) * cD + h_ * cHD + n] = (__bf16)acc[t][r];
    }
}

// ---------------------------------------------------------------------------
// Host-side orchestration
// ---------------------------------------------------------------------------
extern "C" void kernel_launch(void* const* d_in, const int* in_sizes, int n_in,
                              void* d_out, int out_size, void* d_ws,
                              size_t ws_size, hipStream_t stream) {
  const float* x  = (const float*)d_in[0];
  const float* wq = (const float*)d_in[1];
  const float* bq = (const float*)d_in[2];
  const float* wk = (const float*)d_in[3];
  const float* bk = (const float*)d_in[4];
  const float* wv = (const float*)d_in[5];
  const float* bv = (const float*)d_in[6];
  const float* wo = (const float*)d_in[7];
  const float* bo = (const float*)d_in[8];
  const float* w1 = (const float*)d_in[9];
  const float* b1 = (const float*)d_in[10];
  const float* w2 = (const float*)d_in[11];
  const float* b2 = (const float*)d_in[12];
  const float* g1 = (const float*)d_in[13];
  const float* be1 = (const float*)d_in[14];
  const float* g2 = (const float*)d_in[15];
  const float* be2 = (const float*)d_in[16];

  float* out_f = (float*)d_out;   // [B,S,D]
  float* attn = out_f + cM * cD;  // [B,H,S,S]

  char* ws = (char*)d_ws;
  size_t cur = 0;
  auto alloc = [&](size_t bytes) {
    void* p = ws + cur;
    cur = (cur + bytes + 255) & ~(size_t)255;
    return p;
  };
  __bf16* xn_bf  = (__bf16*)alloc(cM * cD * 2);
  __bf16* q_bf   = (__bf16*)alloc(cM * cD * 2);  // [B,H,S,HD]
  __bf16* k_bf   = (__bf16*)alloc(cM * cD * 2);  // [B,H,S,HD]
  __bf16* vt_bf  = (__bf16*)alloc(cM * cD * 2);  // [B,H,HD,S]
  __bf16* o_bf   = (__bf16*)alloc(cM * cD * 2);  // [B,S,D]
  __bf16* xn2_bf = (__bf16*)alloc(cM * cD * 2);
  float*  x_attn = (float*)alloc(cM * cD * 4);
  __bf16* h_bf   = (__bf16*)alloc(cM * cFF * 2);
  __bf16* wqt = (__bf16*)alloc((size_t)cD * cD * 2);   // all weights stored W^T
  __bf16* wkt = (__bf16*)alloc((size_t)cD * cD * 2);
  __bf16* wvt = (__bf16*)alloc((size_t)cD * cD * 2);
  __bf16* wot = (__bf16*)alloc((size_t)cD * cD * 2);
  __bf16* w1t = (__bf16*)alloc((size_t)cD * cFF * 2);  // [FF][D]
  __bf16* w2t = (__bf16*)alloc((size_t)cFF * cD * 2);  // [D][FF]
  (void)ws_size; (void)in_sizes; (void)n_in; (void)out_size;

  // 1) weights -> bf16, transposed to [N][K]
  dim3 gT(cD / 32, cD / 32);
  cast_transpose_kernel<<<gT, 256, 0, stream>>>(wq, wqt, cD, cD);
  cast_transpose_kernel<<<gT, 256, 0, stream>>>(wk, wkt, cD, cD);
  cast_transpose_kernel<<<gT, 256, 0, stream>>>(wv, wvt, cD, cD);
  cast_transpose_kernel<<<gT, 256, 0, stream>>>(wo, wot, cD, cD);
  cast_transpose_kernel<<<dim3(cFF / 32, cD / 32), 256, 0, stream>>>(w1, w1t, cD, cFF);
  cast_transpose_kernel<<<dim3(cD / 32, cFF / 32), 256, 0, stream>>>(w2, w2t, cFF, cD);

  // 2) LN1
  layernorm_kernel<<<cM, 256, 0, stream>>>(x, g1, be1, xn_bf);

  // 3) QKV projections (Q,K -> [B,H,S,HD]; V -> [B,H,HD,S])
  dim3 gP(cD / 128, cM / 128);
  gemm_bf16_kernel<false, 1><<<gP, 256, 0, stream>>>(
      xn_bf, wqt, bq, nullptr, q_bf, nullptr, (int)cM, cD, cD);
  gemm_bf16_kernel<false, 1><<<gP, 256, 0, stream>>>(
      xn_bf, wkt, bk, nullptr, k_bf, nullptr, (int)cM, cD, cD);
  gemm_bf16_kernel<false, 2><<<gP, 256, 0, stream>>>(
      xn_bf, wvt, bv, nullptr, vt_bf, nullptr, (int)cM, cD, cD);

  // 4) scores + causal mask (f32 straight into d_out's attn region)
  dim3 gS(cS / 64, cS / 128, cB * cH);
  attn_scores_kernel<<<gS, 256, 0, stream>>>(q_bf, k_bf, attn);

  // 5) softmax (in place; this IS the attn output)
  softmax_kernel<<<(uint32_t)(cB * cH * cS), 256, 0, stream>>>(attn);

  // 6) attn @ V -> o in [B,S,D] bf16
  dim3 gV(1, cS / 128, cB * cH);
  attn_v_kernel<<<gV, 256, 0, stream>>>(attn, vt_bf, o_bf);

  // 7) O projection + residual(x) -> x_attn (f32)
  gemm_bf16_kernel<false, 0><<<gP, 256, 0, stream>>>(
      o_bf, wot, bo, x, nullptr, x_attn, (int)cM, cD, cD);

  // 8) LN2
  layernorm_kernel<<<cM, 256, 0, stream>>>(x_attn, g2, be2, xn2_bf);

  // 9) FFN1 + GELU -> h bf16
  dim3 gF1(cFF / 128, cM / 128);
  gemm_bf16_kernel<true, 0><<<gF1, 256, 0, stream>>>(
      xn2_bf, w1t, b1, nullptr, h_bf, nullptr, (int)cM, cFF, cD);

  // 10) FFN2 + residual(x_attn) -> out (f32, d_out)
  gemm_bf16_kernel<false, 0><<<gP, 256, 0, stream>>>(
      h_bf, w2t, b2, x_attn, nullptr, out_f, (int)cM, cD, cFF);
}
